// graph_structure_learner_9552007266922
// MI455X (gfx1250) — matrix operand order repaired and checked by reading, not compiled, not run on hardware
//
#include <hip/hip_runtime.h>
#include <hip/hip_bf16.h>

typedef __attribute__((ext_vector_type(16))) __bf16 v16bf;
typedef __attribute__((ext_vector_type(8)))  __bf16 v8bf;
typedef __attribute__((ext_vector_type(8)))  float  v8f;

#define DF      128   // emb dim
#define RF      32    // rel emb dim
#define KF      160   // D + R
#define HF      128   // hidden
#define NSTEP   5     // K steps of 32
#define NT      8     // N tiles of 16
#define BN_EPS  1e-5f
#define SLOPE   0.01f
#define LAMDA   0.5f
#define THRESH  1e-4f

static __device__ __forceinline__ unsigned short f2bf(float f) {
    unsigned u = __float_as_uint(f);
    unsigned r = 0x7FFFu + ((u >> 16) & 1u);   // round-to-nearest-even
    return (unsigned short)((u + r) >> 16);
}
static __device__ __forceinline__ unsigned pack2bf(float a, float b) {
    return (unsigned)f2bf(a) | ((unsigned)f2bf(b) << 16);
}

// order-preserving uint encoding of float (for atomicMax segment max)
static __device__ __forceinline__ unsigned encf(float f) {
    unsigned u = __float_as_uint(f);
    return (u & 0x80000000u) ? ~u : (u | 0x80000000u);
}
static __device__ __forceinline__ float decf(unsigned u) {
    unsigned b = (u & 0x80000000u) ? (u & 0x7FFFFFFFu) : ~u;
    return __uint_as_float(b);
}

// One-time: W0 (row-major [K=160][N=128] fp32) -> column-major bf16 pack: wpack[n*160+k]
__global__ void k_packW0(const float* __restrict__ W0, unsigned short* __restrict__ wpack)
{
    int i = blockIdx.x * blockDim.x + threadIdx.x;   // i = n*KF + k
    if (i < KF * HF) {
        int n = i / KF;
        int k = i - n * KF;
        wpack[i] = f2bf(W0[k * HF + n]);
    }
}

// Per-block: straight 40KB vector copy of prepacked bf16 W0 into LDS
static __device__ __forceinline__ void stage_W0p(const unsigned short* __restrict__ wpack,
                                                 __bf16* Wlds)
{
    const uint4* src = (const uint4*)wpack;
    uint4* dst = (uint4*)Wlds;
    for (int i = threadIdx.x; i < (KF * HF) / 8; i += blockDim.x)
        dst[i] = src[i];
}

// One wave computes h = x @ W0 for its 16-edge tile into acc[NT] (fp32 16x16 C tiles).
// Lane's channel within tile nt is (nt*16 + (lane&15)); its 8 values cover edges
// m = r + 8*(lane>>4), r = 0..7.
// NO block barriers inside: the A tile is private to this wave, and LDS ops from a
// single wave are executed in order (DScnt); the compiler's s_wait_dscnt covers the
// store->load dependence.
static __device__ __forceinline__ void gemm_tile(
    const float* __restrict__ n_feat, const float* __restrict__ rel_emb,
    const int* __restrict__ row, const int* __restrict__ col,
    const int* __restrict__ etype, int E, int base,
    const __bf16* Wlds, __bf16* aw, int lane, v8f acc[NT])
{
    int eSt = base + (lane >> 1);
    int eCl = eSt < E ? eSt : (E - 1);          // clamp tail: duplicate last edge (masked later)
    int r0 = row[eCl], c0 = col[eCl], t0 = etype[eCl];
    int er   = lane >> 1;                        // edge row this lane stages
    int sub  = (lane & 1) * 16;                  // which 16 of 32 K values
    int m    = lane & 15;                        // fragment M (A) / N (B)
    int half = lane >> 4;

    #pragma unroll
    for (int nt = 0; nt < NT; ++nt)
        acc[nt] = (v8f){0.f, 0.f, 0.f, 0.f, 0.f, 0.f, 0.f, 0.f};

    #pragma unroll
    for (int ks = 0; ks < NSTEP; ++ks) {
        int k0 = ks * 32;
        // ---- stage A tile (16 edges x 32 bf16), packed b64 stores ----
        if (k0 < DF) {
            const float4* pr = (const float4*)(n_feat + (size_t)r0 * DF + k0 + sub);
            const float4* pc = (const float4*)(n_feat + (size_t)c0 * DF + k0 + sub);
            #pragma unroll
            for (int j = 0; j < 4; ++j) {
                float4 a = pr[j], b = pc[j];
                uint2 v;
                v.x = pack2bf(__expf(-fabsf(a.x - b.x)), __expf(-fabsf(a.y - b.y)));
                v.y = pack2bf(__expf(-fabsf(a.z - b.z)), __expf(-fabsf(a.w - b.w)));
                *(uint2*)(aw + er * 32 + sub + j * 4) = v;
            }
        } else {
            const float4* pr = (const float4*)(rel_emb + (size_t)t0 * RF + sub);
            #pragma unroll
            for (int j = 0; j < 4; ++j) {
                float4 a = pr[j];
                uint2 v;
                v.x = pack2bf(a.x, a.y);
                v.y = pack2bf(a.z, a.w);
                *(uint2*)(aw + er * 32 + sub + j * 4) = v;
            }
        }
        // ---- A fragment: lane m, 16 contiguous K values [half*16, +16) ----
        union { v16bf v; v8bf h[2]; } af;
        af.h[0] = *(const v8bf*)(aw + m * 32 + half * 16);
        af.h[1] = *(const v8bf*)(aw + m * 32 + half * 16 + 8);
        // ---- 8 N tiles: B frag = column n, 16 contiguous K values ----
        #pragma unroll
        for (int nt = 0; nt < NT; ++nt) {
            int n = nt * 16 + m;
            v16bf bf = *(const v16bf*)(Wlds + n * KF + k0 + half * 16);
            acc[nt] = __builtin_amdgcn_wmma_f32_16x16x32_bf16(
                false, af.v, false, bf, (short)0, acc[nt], false, false);
        }
    }
}

// -------- pass 1: GEMM + per-channel sum / sumsq (for BatchNorm stats) --------
__global__ __launch_bounds__(256) void k_gemm_stats(
    const float* __restrict__ n_feat, const float* __restrict__ rel_emb,
    const unsigned short* __restrict__ wpack, const float* __restrict__ b0,
    const int* __restrict__ row, const int* __restrict__ col,
    const int* __restrict__ etype, int E,
    float* __restrict__ gsum, float* __restrict__ gsq)
{
    __shared__ __align__(32) __bf16 Wlds[HF * KF];
    __shared__ __align__(32) __bf16 Atile[8 * 16 * 32];
    __shared__ float bsum[HF], bsq[HF];
    int tid = threadIdx.x;
    if (tid < HF) { bsum[tid] = 0.f; bsq[tid] = 0.f; }
    stage_W0p(wpack, Wlds);
    __syncthreads();

    int wave = tid >> 5, lane = tid & 31;
    int base = blockIdx.x * 128 + wave * 16;
    v8f acc[NT];
    gemm_tile(n_feat, rel_emb, row, col, etype, E, base,
              Wlds, Atile + wave * (16 * 32), lane, acc);

    int m15 = lane & 15, half = lane >> 4;
    #pragma unroll
    for (int nt = 0; nt < NT; ++nt) {
        int c = nt * 16 + m15;
        float bb = b0[c];
        float s = 0.f, q = 0.f;
        #pragma unroll
        for (int r = 0; r < 8; ++r) {
            int e = base + r + 8 * half;
            if (e < E) {
                float h = acc[nt][r] + bb;
                s += h; q += h * h;
            }
        }
        atomicAdd(&bsum[c], s);
        atomicAdd(&bsq[c], q);
    }
    __syncthreads();
    if (tid < HF) {
        atomicAdd(&gsum[tid], bsum[tid]);
        atomicAdd(&gsq[tid],  bsq[tid]);
    }
}

// -------- BN stats -> scale/shift --------
__global__ void k_bnstats(const float* __restrict__ gsum, const float* __restrict__ gsq,
                          const float* __restrict__ gamma, const float* __restrict__ beta,
                          float invE, float* __restrict__ scale, float* __restrict__ shift)
{
    int c = threadIdx.x;
    if (c < HF) {
        float mu   = gsum[c] * invE;
        float var  = gsq[c] * invE - mu * mu;
        float rstd = rsqrtf(var + BN_EPS);
        float sc   = gamma[c] * rstd;
        scale[c] = sc;
        shift[c] = beta[c] - mu * sc;
    }
}

// -------- pass 2: recompute GEMM, apply BN + LeakyReLU + W1 dot -> w[e] --------
__global__ __launch_bounds__(256) void k_gemm_w(
    const float* __restrict__ n_feat, const float* __restrict__ rel_emb,
    const unsigned short* __restrict__ wpack, const float* __restrict__ b0,
    const int* __restrict__ row, const int* __restrict__ col,
    const int* __restrict__ etype, int E,
    const float* __restrict__ scale, const float* __restrict__ shift,
    const float* __restrict__ W1, const float* __restrict__ b1,
    float* __restrict__ wout)
{
    __shared__ __align__(32) __bf16 Wlds[HF * KF];
    __shared__ __align__(32) __bf16 Atile[8 * 16 * 32];
    stage_W0p(wpack, Wlds);
    __syncthreads();

    int tid = threadIdx.x;
    int wave = tid >> 5, lane = tid & 31;
    int base = blockIdx.x * 128 + wave * 16;
    v8f acc[NT];
    gemm_tile(n_feat, rel_emb, row, col, etype, E, base,
              Wlds, Atile + wave * (16 * 32), lane, acc);

    int m15 = lane & 15, half = lane >> 4;
    float part[8];
    #pragma unroll
    for (int r = 0; r < 8; ++r) part[r] = 0.f;
    #pragma unroll
    for (int nt = 0; nt < NT; ++nt) {
        int c = nt * 16 + m15;
        float sc = scale[c], sh = shift[c], w1 = W1[c], bb = b0[c];
        #pragma unroll
        for (int r = 0; r < 8; ++r) {
            float h = acc[nt][r] + bb;
            float t = h * sc + sh;
            t = (t >= 0.f) ? t : SLOPE * t;
            part[r] += t * w1;
        }
    }
    // reduce over the 16 lanes of each half-wave (each holds 8 of the 128 channels)
    #pragma unroll
    for (int off = 1; off < 16; off <<= 1)
        #pragma unroll
        for (int r = 0; r < 8; ++r)
            part[r] += __shfl_xor(part[r], off, 32);
    if (m15 == 0) {
        float bb1 = b1[0];
        #pragma unroll
        for (int r = 0; r < 8; ++r) {
            int e = base + r + 8 * half;
            if (e < E) wout[e] = part[r] + bb1;
        }
    }
}

// -------- elementwise / segment kernels --------
__global__ void k_init(int N, float* gsum, float* gsq, unsigned* menc, float* sbuf)
{
    int i = blockIdx.x * blockDim.x + threadIdx.x;
    if (i < 2 * HF) { gsum[i] = 0.f; }   // gsum+gsq contiguous (gsq = gsum+HF)
    if (i < N) { menc[i] = 0u; sbuf[i] = 0.f; }
    (void)gsq;
}

__global__ void k_blend(const int* __restrict__ ori, int EO, float* __restrict__ w)
{
    int i = blockIdx.x * blockDim.x + threadIdx.x;
    if (i < EO) {
        int e = ori[i];
        w[e] = (1.f - LAMDA) * w[e] + LAMDA;
    }
}

__global__ void k_segmax(const float* __restrict__ w, const int* __restrict__ col,
                         int E, unsigned* __restrict__ menc)
{
    int i = blockIdx.x * blockDim.x + threadIdx.x;
    if (i < E) atomicMax(&menc[col[i]], encf(w[i]));
}

__global__ void k_expsum(const float* __restrict__ w, const int* __restrict__ col, int E,
                         const unsigned* __restrict__ menc,
                         float* __restrict__ ebuf, float* __restrict__ sbuf)
{
    int i = blockIdx.x * blockDim.x + threadIdx.x;
    if (i < E) {
        int c = col[i];
        float e = __expf(w[i] - decf(menc[c]));
        ebuf[i] = e;
        atomicAdd(&sbuf[c], e);
    }
}

__global__ void k_norm(const float* __restrict__ ebuf, const int* __restrict__ col, int E,
                       const float* __restrict__ sbuf, float* __restrict__ out)
{
    int i = blockIdx.x * blockDim.x + threadIdx.x;
    if (i < E) {
        float o = ebuf[i] / sbuf[col[i]];
        out[i] = (o > THRESH) ? o : 0.f;
    }
}

extern "C" void kernel_launch(void* const* d_in, const int* in_sizes, int n_in,
                              void* d_out, int out_size, void* d_ws, size_t ws_size,
                              hipStream_t stream)
{
    const float* n_feat = (const float*)d_in[0];
    const float* rel    = (const float*)d_in[1];
    const float* W0     = (const float*)d_in[2];
    const float* b0     = (const float*)d_in[3];
    const float* gamma  = (const float*)d_in[4];
    const float* beta   = (const float*)d_in[5];
    const float* W1     = (const float*)d_in[6];
    const float* b1     = (const float*)d_in[7];
    const int*   row    = (const int*)d_in[8];
    const int*   col    = (const int*)d_in[9];
    const int*   etype  = (const int*)d_in[10];
    const int*   ori    = (const int*)d_in[11];
    int E  = in_sizes[8];
    int N  = in_sizes[0] / DF;
    int EO = in_sizes[11];

    float* ws             = (float*)d_ws;
    unsigned short* wpack = (unsigned short*)ws;        // KF*HF bf16 = 10240 floats
    float* gsum    = ws + 10240;                        // HF
    float* gsq     = gsum + HF;                         // HF
    float* scale   = gsq + HF;                          // HF
    float* shift   = scale + HF;                        // HF
    float* wbuf    = shift + HF;                        // E
    float* ebuf    = wbuf + E;                          // E
    unsigned* menc = (unsigned*)(ebuf + E);             // N
    float* sbuf    = (float*)(menc + N);                // N
    float* out     = (float*)d_out;

    const int TB = 256;
    int initBlocks = (N > 2 * HF ? N : 2 * HF);
    k_init<<<(initBlocks + TB - 1) / TB, TB, 0, stream>>>(N, gsum, gsq, menc, sbuf);
    k_packW0<<<(KF * HF + TB - 1) / TB, TB, 0, stream>>>(W0, wpack);

    int gemmBlocks = (E + 127) / 128;
    k_gemm_stats<<<gemmBlocks, TB, 0, stream>>>(n_feat, rel, wpack, b0, row, col, etype,
                                                E, gsum, gsq);
    k_bnstats<<<1, HF, 0, stream>>>(gsum, gsq, gamma, beta, 1.0f / (float)E, scale, shift);
    k_gemm_w<<<gemmBlocks, TB, 0, stream>>>(n_feat, rel, wpack, b0, row, col, etype, E,
                                            scale, shift, W1, b1, wbuf);
    k_blend<<<(EO + TB - 1) / TB, TB, 0, stream>>>(ori, EO, wbuf);
    k_segmax<<<(E + TB - 1) / TB, TB, 0, stream>>>(wbuf, col, E, menc);
    k_expsum<<<(E + TB - 1) / TB, TB, 0, stream>>>(wbuf, col, E, menc, ebuf, sbuf);
    k_norm<<<(E + TB - 1) / TB, TB, 0, stream>>>(ebuf, col, E, sbuf, out);
}
